// HybridQLSTM_65481071400970
// MI455X (gfx1250) — matrix operand order repaired and checked by reading, not compiled, not run on hardware
//
#include <hip/hip_runtime.h>

// ---------------------------------------------------------------------------
// HybridQLSTM on gfx1250: per-step fused (GEMM + kernel-factor + gates) using
// v_wmma_f32_16x16x32_bf16, weights pre-packed into WMMA B-fragment order.
// T=512, B=256, D=256, H=512, D+H=768, 4H=2048.
// ---------------------------------------------------------------------------

typedef __attribute__((ext_vector_type(16))) __bf16 v16bf;
typedef __attribute__((ext_vector_type(8)))  __bf16 v8bf;
typedef __attribute__((ext_vector_type(8)))  float  v8f;

#define TT 512
#define BB 256
#define DD 256
#define HH 512
#define KK 768          // D+H
#define NN 2048         // 4H
#define KCHUNKS 24      // 768 / 32
#define TILE_HALVES (32*16)          // halves per (ntile,kc) fragment chunk
#define NTILE_STRIDE (KCHUNKS*TILE_HALVES)  // 12288 halves per n-tile

// ---------------------------------------------------------------------------
// One-time: repack [Wf|Wi|Wg|Wo] (each 768x512 f32, row-major) into bf16 in
// WMMA B-fragment order:  Wp[((ntile*24 + kc)*32 + lane)*16 + j]
//   lane: n = ntile*16 + (lane&15), kbase = kc*32 + (lane>>4)*16
//   half j holds K = kbase + j   (VGPR v = j/2 holds K=2v (lo), 2v+1 (hi))
// ---------------------------------------------------------------------------
__global__ __launch_bounds__(256)
void repack_w_kernel(const float* __restrict__ Wf, const float* __restrict__ Wi,
                     const float* __restrict__ Wg, const float* __restrict__ Wo,
                     __bf16* __restrict__ Wp)
{
    int p = blockIdx.x * 256 + threadIdx.x;          // 0 .. 768*2048-1
    if (p >= KK * NN) return;
    int j     = p & 15;
    int l     = (p >> 4) & 31;
    int kc    = (p >> 9) % KCHUNKS;
    int ntile = (p >> 9) / KCHUNKS;
    int k = kc * 32 + (l >> 4) * 16 + j;
    int n = ntile * 16 + (l & 15);
    int gate = n >> 9;                                // 0..3
    int col  = n & 511;
    const float* Wsrc = (gate == 0) ? Wf : (gate == 1) ? Wi : (gate == 2) ? Wg : Wo;
    Wp[p] = (__bf16)Wsrc[k * HH + col];
}

__global__ __launch_bounds__(256)
void zero_f32_kernel(float* __restrict__ a, int n)
{
    int i = blockIdx.x * 256 + threadIdx.x;
    if (i < n) a[i] = 0.0f;
}

// ---------------------------------------------------------------------------
// One recurrent step. Grid (8,8), 256 threads (8 waves).
// Block (bm,bj): batch rows [bm*32, bm*32+32), hidden cols [bj*64, bj*64+64),
// computing all 4 gates for those hidden cols.
// wave w: row-tile mt = w&1, gate = w>>1; 4 WMMA n-tiles (64 cols).
// ---------------------------------------------------------------------------
__global__ __launch_bounds__(256)
void qlstm_step_kernel(const float* __restrict__ x_t,     // (256,256)
                       const float* __restrict__ h_in,    // (256,512)
                       float* __restrict__ h_out,         // (256,512)
                       float* __restrict__ c,             // (256,512)
                       float* __restrict__ out_t,         // (256,512)
                       const __bf16* __restrict__ Wp,     // packed bf16 weights
                       const float* __restrict__ bfv, const float* __restrict__ biv,
                       const float* __restrict__ bgv, const float* __restrict__ bov,
                       const float* __restrict__ center,  // (768)
                       const float* __restrict__ gammap)  // scalar
{
    // 48KB buffer: phase 1-2 holds comb tile (32x768 bf16);
    // phase 3-4 aliased as z tiles (4 gates x 32 rows x 64 cols f32 = 32KB).
    __shared__ __align__(16) unsigned char smem[32 * KK * 2];
    __bf16* sComb = (__bf16*)smem;
    float*  sZ    = (float*)smem;
    __shared__ float sK[32];

    const int tid = threadIdx.x;
    const int bm  = blockIdx.x;      // 0..7
    const int bj  = blockIdx.y;      // 0..7
    const float gamma = gammap[0];

    if (tid < 32) sK[tid] = 0.0f;
    __syncthreads();

    // ---- Phase 1: stage comb tile in LDS (bf16) + accumulate ||comb-center||^2
    {
        const int r  = tid >> 3;                  // row 0..31
        const int tc = tid & 7;                   // 8 threads per row
        const int mg = bm * 32 + r;
        const int c0 = tc * 96;
        float ssum = 0.0f;
        #pragma unroll 4
        for (int cc = 0; cc < 96; ++cc) {
            int col = c0 + cc;
            float v = (col < DD) ? x_t[mg * DD + col]
                                 : h_in[mg * HH + (col - DD)];
            float d = v - center[col];
            ssum += d * d;
            sComb[r * KK + col] = (__bf16)v;
        }
        atomicAdd(&sK[r], ssum);
    }
    __syncthreads();

    // ---- Phase 2: WMMA GEMM over K=768 (24 chunks of 32)
    const int wave = tid >> 5;            // 0..7
    const int lane = tid & 31;
    const int mt   = wave & 1;            // row tile
    const int gate = wave >> 1;           // 0..3
    const int arow = mt * 16 + (lane & 15);
    const int kg   = lane >> 4;           // K-half group for A fragment

    v8f acc0 = {}; v8f acc1 = {}; v8f acc2 = {}; v8f acc3 = {};
    const int ntg0 = gate * 32 + bj * 4;  // first of 4 global n-tiles (128 total)
    const __bf16* Wbase = Wp + (size_t)ntg0 * NTILE_STRIDE + (size_t)lane * 16;

    for (int kc = 0; kc < KCHUNKS; ++kc) {
        // A fragment (contiguous K order, matching packed B): K = kg*16 + j
        const __bf16* ap = &sComb[arow * KK + kc * 32 + kg * 16];
        v8bf alo = *(const v8bf*)(ap);
        v8bf ahi = *(const v8bf*)(ap + 8);
        v16bf a;
        #pragma unroll
        for (int i = 0; i < 8; ++i) { a[i] = alo[i]; a[8 + i] = ahi[i]; }

        // B fragments: contiguous 32B per lane from packed weights
        const __bf16* bp = Wbase + (size_t)kc * TILE_HALVES;
        v16bf b0 = *(const v16bf*)(bp);
        v16bf b1 = *(const v16bf*)(bp + 1 * NTILE_STRIDE);
        v16bf b2 = *(const v16bf*)(bp + 2 * NTILE_STRIDE);
        v16bf b3 = *(const v16bf*)(bp + 3 * NTILE_STRIDE);

        acc0 = __builtin_amdgcn_wmma_f32_16x16x32_bf16(false, a, false, b0, (short)0, acc0, false, false);
        acc1 = __builtin_amdgcn_wmma_f32_16x16x32_bf16(false, a, false, b1, (short)0, acc1, false, false);
        acc2 = __builtin_amdgcn_wmma_f32_16x16x32_bf16(false, a, false, b2, (short)0, acc2, false, false);
        acc3 = __builtin_amdgcn_wmma_f32_16x16x32_bf16(false, a, false, b3, (short)0, acc3, false, false);
    }
    __syncthreads();   // all waves done reading sComb -> safe to alias as sZ

    // ---- Phase 3: spill z tiles. C/D layout: VGPR r -> M = r + 8*(lane>>4)
    {
        const int colb  = lane & 15;
        const int rowhi = 8 * (lane >> 4);
        #pragma unroll
        for (int r8 = 0; r8 < 8; ++r8) {
            int rr = mt * 16 + rowhi + r8;
            float* zr = &sZ[(gate * 32 + rr) * 64];
            zr[0 * 16 + colb] = acc0[r8];
            zr[1 * 16 + colb] = acc1[r8];
            zr[2 * 16 + colb] = acc2[r8];
            zr[3 * 16 + colb] = acc3[r8];
        }
    }
    __syncthreads();

    // ---- Phase 4: gates + state update (32x64 elements, 8 per thread)
    #pragma unroll
    for (int e = 0; e < 8; ++e) {
        int idx = tid + e * 256;          // 0..2047
        int r   = idx >> 6;               // 0..31
        int jj  = idx & 63;
        int mg  = bm * 32 + r;
        int jg  = bj * 64 + jj;
        float kv = expf(-gamma * sK[r]);
        float zf = sZ[(0 * 32 + r) * 64 + jj] + bfv[jg];
        float zi = sZ[(1 * 32 + r) * 64 + jj] + biv[jg];
        float zg = sZ[(2 * 32 + r) * 64 + jj] + bgv[jg];
        float zo = sZ[(3 * 32 + r) * 64 + jj] + bov[jg];
        float f = 1.0f / (1.0f + expf(-zf * kv));
        float i = 1.0f / (1.0f + expf(-zi * kv));
        float g = tanhf(zg);
        float o = 1.0f / (1.0f + expf(-zo * kv));
        size_t off = (size_t)mg * HH + jg;
        float cnew = f * c[off] + i * g;
        c[off] = cnew;
        float hnew = o * tanhf(cnew);
        h_out[off] = hnew;
        out_t[off] = hnew;
    }
}

__global__ __launch_bounds__(256)
void copy_final_kernel(const float* __restrict__ hfin, const float* __restrict__ cfin,
                       float* __restrict__ dst)
{
    int i = blockIdx.x * 256 + threadIdx.x;
    if (i < BB * HH)            dst[i] = hfin[i];
    else if (i < 2 * BB * HH)   dst[i] = cfin[i - BB * HH];
}

// ---------------------------------------------------------------------------
extern "C" void kernel_launch(void* const* d_in, const int* in_sizes, int n_in,
                              void* d_out, int out_size, void* d_ws, size_t ws_size,
                              hipStream_t stream)
{
    (void)in_sizes; (void)n_in; (void)out_size; (void)ws_size;

    const float* inputs = (const float*)d_in[0];   // (T,B,D)
    const float* Wf     = (const float*)d_in[1];
    const float* bfv    = (const float*)d_in[2];
    const float* Wi     = (const float*)d_in[3];
    const float* biv    = (const float*)d_in[4];
    const float* Wg     = (const float*)d_in[5];
    const float* bgv    = (const float*)d_in[6];
    const float* Wo     = (const float*)d_in[7];
    const float* bov    = (const float*)d_in[8];
    const float* gammap = (const float*)d_in[9];
    const float* center = (const float*)d_in[10];
    float* out = (float*)d_out;

    // Workspace layout (bytes):
    //   Wp   @ 0         : 768*2048*2 = 3,145,728
    //   h[0] @ 3,145,728 : 256*512*4  =   524,288
    //   h[1] @ 3,670,016 : 524,288
    //   c    @ 4,194,304 : 524,288      (total 4,718,592)
    char* ws = (char*)d_ws;
    __bf16* Wp    = (__bf16*)(ws);
    float*  hbuf0 = (float*)(ws + 3145728);
    float*  hbuf1 = (float*)(ws + 3670016);
    float*  cbuf  = (float*)(ws + 4194304);
    float*  hbuf[2] = { hbuf0, hbuf1 };

    // 1) pack weights into WMMA B-fragment bf16 layout
    repack_w_kernel<<<(KK * NN) / 256, 256, 0, stream>>>(Wf, Wi, Wg, Wo, Wp);
    // 2) zero initial state
    zero_f32_kernel<<<(BB * HH + 255) / 256, 256, 0, stream>>>(hbuf[0], BB * HH);
    zero_f32_kernel<<<(BB * HH + 255) / 256, 256, 0, stream>>>(cbuf, BB * HH);

    // 3) recurrence: one fused kernel per step (double-buffered h)
    for (int t = 0; t < TT; ++t) {
        const float* x_t  = inputs + (size_t)t * BB * DD;
        float*       outt = out    + (size_t)t * BB * HH;
        qlstm_step_kernel<<<dim3(8, 8), 256, 0, stream>>>(
            x_t, hbuf[t & 1], hbuf[(t + 1) & 1], cbuf, outt,
            Wp, bfv, biv, bgv, bov, center, gammap);
    }

    // 4) append (hx, cx); after t=511 the latest h is in hbuf[0]
    copy_final_kernel<<<(2 * BB * HH + 255) / 256, 256, 0, stream>>>(
        hbuf[0], cbuf, out + (size_t)TT * BB * HH);
}
